// ContrastiveLoss_8821862826751
// MI455X (gfx1250) — matrix operand order repaired and checked by reading, not compiled, not run on hardware
//
#include <hip/hip_runtime.h>
#include <math.h>

typedef __attribute__((ext_vector_type(16))) _Float16 v16h;
typedef __attribute__((ext_vector_type(8)))  _Float16 v8h;
typedef __attribute__((ext_vector_type(4)))  _Float16 v4h;
typedef __attribute__((ext_vector_type(8)))  float    v8f;
typedef __attribute__((ext_vector_type(4)))  int      v4i;

#define NROWS 8192
#define DDIM  1024
#define INV_T (1.0f / 0.07f)
#define EPSN  1e-8f

// CDNA5 async global->LDS copy path (ASYNCcnt); guarded so absence of the
// builtins on this toolchain falls back to the proven sync staging path.
#if __has_builtin(__builtin_amdgcn_global_load_async_to_lds_b128) && \
    __has_builtin(__builtin_amdgcn_s_wait_asynccnt)
#define USE_ASYNC_COPY 1
#else
#define USE_ASYNC_COPY 0
#endif

#if USE_ASYNC_COPY
// exact param types per hipcc diagnostic: (v4i AS1*, v4i AS3*, imm, imm)
typedef __attribute__((address_space(1))) v4i* gas_ptr;   // global <4 x i32>*
typedef __attribute__((address_space(3))) v4i* las_ptr;   // LDS    <4 x i32>*
#endif

// ---------------------------------------------------------------------------
// Kernel 1: row L2-normalize, emit f16 copy (fits in 192MB L2 -> compute bound)
// ---------------------------------------------------------------------------
__global__ void __launch_bounds__(256)
normalize_f16(const float* __restrict__ z, _Float16* __restrict__ zn) {
    const int row = blockIdx.x;
    const int tid = threadIdx.x;
    const float4 v = *(const float4*)(z + (size_t)row * DDIM + tid * 4);
    float ss = v.x * v.x + v.y * v.y + v.z * v.z + v.w * v.w;
    for (int off = 16; off > 0; off >>= 1) ss += __shfl_down(ss, off, 32);
    __shared__ float wsum[8];
    const int wid = tid >> 5, lane = tid & 31;
    if (lane == 0) wsum[wid] = ss;
    __syncthreads();
    const float tot = wsum[0] + wsum[1] + wsum[2] + wsum[3] +
                      wsum[4] + wsum[5] + wsum[6] + wsum[7];
    const float inv = 1.0f / fmaxf(sqrtf(tot), EPSN);
    v4h o;
    o[0] = (_Float16)(v.x * inv);
    o[1] = (_Float16)(v.y * inv);
    o[2] = (_Float16)(v.z * inv);
    o[3] = (_Float16)(v.w * inv);
    *(v4h*)(zn + (size_t)row * DDIM + tid * 4) = o;
}

// ---------------------------------------------------------------------------
// Kernel 2: positive-pair logits: pos[i] = dot(zn[i], zn[i^4096]) / T
// ---------------------------------------------------------------------------
__global__ void __launch_bounds__(256)
pos_kernel(const _Float16* __restrict__ zn, float* __restrict__ pos) {
    const int i = blockIdx.x;
    const int j = i ^ (NROWS / 2);   // (i - N/2) mod N == i ^ (N/2) for N=8192
    const int tid = threadIdx.x;
    const v4h a = *(const v4h*)(zn + (size_t)i * DDIM + tid * 4);
    const v4h b = *(const v4h*)(zn + (size_t)j * DDIM + tid * 4);
    float d = (float)a[0] * (float)b[0] + (float)a[1] * (float)b[1] +
              (float)a[2] * (float)b[2] + (float)a[3] * (float)b[3];
    for (int off = 16; off > 0; off >>= 1) d += __shfl_down(d, off, 32);
    __shared__ float wsum[8];
    const int wid = tid >> 5, lane = tid & 31;
    if (lane == 0) wsum[wid] = d;
    __syncthreads();
    if (tid == 0) {
        float tot = wsum[0] + wsum[1] + wsum[2] + wsum[3] +
                    wsum[4] + wsum[5] + wsum[6] + wsum[7];
        pos[i] = tot * INV_T;
    }
}

// ---------------------------------------------------------------------------
// Kernel 3: fused zn@zn^T GEMM (WMMA f16->f32) + masked exp + row-sum.
// Block = 128 rows x 1024 cols (8 col-tiles of 128). Wave (8 per block,
// arranged 4M x 2N) owns a 32x64 output patch = 2x4 WMMA accumulators.
// No max-shift needed: sim/T <= 14.29 so sums stay in f32 range.
// ---------------------------------------------------------------------------
__global__ void __launch_bounds__(256)
simlse_kernel(const _Float16* __restrict__ zn, float* __restrict__ part) {
    constexpr int BM = 128, BN = 128, BK = 32, LDSTR = 40;  // padded stride
    __shared__ alignas(16) _Float16 lA[BM * LDSTR];
    __shared__ alignas(16) _Float16 lB[BN * LDSTR];
    __shared__ float rowsum[BM];

    const int tid = threadIdx.x;
    const int rowbase = blockIdx.x * BM;
    const int cs = blockIdx.y;                 // column split 0..7 (1024 cols each)
    if (tid < BM) rowsum[tid] = 0.0f;

    const int wid = tid >> 5, lane = tid & 31;
    const int wm = wid & 3;                    // wave M position (4 x 32 rows)
    const int wn = wid >> 2;                   // wave N position (2 x 64 cols)
    const int hi = lane >> 4, ln = lane & 15;

    // global->LDS copy coordinates (each thread: 32B of A + 32B of B)
    const int cr = tid >> 1;
    const int ko = (tid & 1) * 16;

    for (int cb = 0; cb < 8; ++cb) {
        const int colbase = cs * 1024 + cb * BN;
        v8f acc[2][4] = {};

        for (int kb = 0; kb < DDIM; kb += BK) {
            __syncthreads();   // previous fragments consumed
            {
                const _Float16* ga = zn + (size_t)(rowbase + cr) * DDIM + kb + ko;
                const _Float16* gb = zn + (size_t)(colbase + cr) * DDIM + kb + ko;
                if (kb + BK < DDIM) {   // global_prefetch_b8 for next K slice
                    __builtin_prefetch((const char*)ga + BK * 2, 0, 3);
                    __builtin_prefetch((const char*)gb + BK * 2, 0, 3);
                }
#if USE_ASYNC_COPY
                // CDNA5 async copy: GLOBAL_LOAD_ASYNC_TO_LDS_B128 (ASYNCcnt)
                __builtin_amdgcn_global_load_async_to_lds_b128(
                    (gas_ptr)ga, (las_ptr)&lA[cr * LDSTR + ko], 0, 0);
                __builtin_amdgcn_global_load_async_to_lds_b128(
                    (gas_ptr)(ga + 8), (las_ptr)&lA[cr * LDSTR + ko + 8], 0, 0);
                __builtin_amdgcn_global_load_async_to_lds_b128(
                    (gas_ptr)gb, (las_ptr)&lB[cr * LDSTR + ko], 0, 0);
                __builtin_amdgcn_global_load_async_to_lds_b128(
                    (gas_ptr)(gb + 8), (las_ptr)&lB[cr * LDSTR + ko + 8], 0, 0);
                __builtin_amdgcn_s_wait_asynccnt(0);
#else
                const float4 a0 = ((const float4*)ga)[0], a1 = ((const float4*)ga)[1];
                const float4 b0 = ((const float4*)gb)[0], b1 = ((const float4*)gb)[1];
                *(float4*)(&lA[cr * LDSTR + ko])     = a0;
                *(float4*)(&lA[cr * LDSTR + ko + 8]) = a1;
                *(float4*)(&lB[cr * LDSTR + ko])     = b0;
                *(float4*)(&lB[cr * LDSTR + ko + 8]) = b1;
#endif
            }
            __syncthreads();

            union { v16h v; v8h h[2]; } af[2], bf[4];
            // A (16x32 f16): lanes 0-15 hold K 0-7 & 16-23, lanes 16-31 K 8-15 & 24-31
            for (int mt = 0; mt < 2; ++mt) {
                const _Float16* p = &lA[(wm * 32 + mt * 16 + ln) * LDSTR + hi * 8];
                af[mt].h[0] = *(const v8h*)(p);
                af[mt].h[1] = *(const v8h*)(p + 16);
            }
            // B (32x16 f16): lanes 0-15 hold K 0-7 & 16-23, lanes 16-31 K 8-15 & 24-31 (col-major src)
            for (int nt = 0; nt < 4; ++nt) {
                const _Float16* p = &lB[(wn * 64 + nt * 16 + ln) * LDSTR + hi * 8];
                bf[nt].h[0] = *(const v8h*)(p);
                bf[nt].h[1] = *(const v8h*)(p + 16);
            }
            for (int mt = 0; mt < 2; ++mt)
                for (int nt = 0; nt < 4; ++nt)
                    acc[mt][nt] = __builtin_amdgcn_wmma_f32_16x16x32_f16(
                        false, af[mt].v, false, bf[nt].v,
                        (short)0, acc[mt][nt], false, false);
        }

        // Epilogue: C layout -> M = j + 8*hi, N = ln within each 16x16 tile.
        for (int mt = 0; mt < 2; ++mt) {
            for (int j = 0; j < 8; ++j) {
                const int rloc = wm * 32 + mt * 16 + j + hi * 8;
                const int rg = rowbase + rloc;
                float s = 0.0f;
                for (int nt = 0; nt < 4; ++nt) {
                    const int cg = colbase + wn * 64 + nt * 16 + ln;
                    const float v = acc[mt][nt][j] * INV_T;
                    s += (rg == cg) ? 0.0f : __expf(v);   // diag masked -> exp(-inf)=0
                }
                // reduce across the 16 column-lanes of this half-wave
                s += __shfl_xor(s, 1, 16);
                s += __shfl_xor(s, 2, 16);
                s += __shfl_xor(s, 4, 16);
                s += __shfl_xor(s, 8, 16);
                if (ln == 0) atomicAdd(&rowsum[rloc], s);  // ds_add_f32
            }
        }
    }
    __syncthreads();
    if (tid < BM)  // one deterministic writer per (split,row): no global atomics
        part[(size_t)cs * NROWS + rowbase + tid] = rowsum[tid];
}

// ---------------------------------------------------------------------------
// Kernel 4: nll mean = mean_i( log(sum_cs part[cs][i]) - pos[i] )
// ---------------------------------------------------------------------------
__global__ void __launch_bounds__(256)
finalize_kernel(const float* __restrict__ part, const float* __restrict__ pos,
                float* __restrict__ out) {
    const int tid = threadIdx.x;
    float a = 0.0f;
    for (int i = tid; i < NROWS; i += 256) {
        float se = 0.0f;
        for (int csr = 0; csr < 8; ++csr) se += part[(size_t)csr * NROWS + i];
        a += logf(se) - pos[i];
    }
    for (int off = 16; off > 0; off >>= 1) a += __shfl_down(a, off, 32);
    __shared__ float wsum[8];
    const int wid = tid >> 5, lane = tid & 31;
    if (lane == 0) wsum[wid] = a;
    __syncthreads();
    if (tid == 0) {
        float tot = wsum[0] + wsum[1] + wsum[2] + wsum[3] +
                    wsum[4] + wsum[5] + wsum[6] + wsum[7];
        out[0] = tot / (float)NROWS;
    }
}

// ---------------------------------------------------------------------------
extern "C" void kernel_launch(void* const* d_in, const int* in_sizes, int n_in,
                              void* d_out, int out_size, void* d_ws, size_t ws_size,
                              hipStream_t stream) {
    (void)in_sizes; (void)n_in; (void)out_size; (void)ws_size;
    const float* z = (const float*)d_in[0];
    float* out = (float*)d_out;

    // workspace layout: zn16 (16MB) | pos (32KB) | part (256KB)
    _Float16* zn = (_Float16*)d_ws;
    float* pos   = (float*)((char*)d_ws + (size_t)NROWS * DDIM * sizeof(_Float16));
    float* part  = pos + NROWS;

    normalize_f16<<<NROWS, 256, 0, stream>>>(z, zn);
    pos_kernel<<<NROWS, 256, 0, stream>>>(zn, pos);
    dim3 grid(NROWS / 128, 8);
    simlse_kernel<<<grid, 256, 0, stream>>>(zn, part);
    finalize_kernel<<<1, 256, 0, stream>>>(part, pos, out);
}
